// GeometricAttention_90537910600028
// MI455X (gfx1250) — compile-verified, matrix operand order checked by reading, not run on hardware
//
#include <hip/hip_runtime.h>
#include <hip/hip_bf16.h>
#include <math.h>

// ---------------------------------------------------------------------------
// GeometricAttention for MI455X (gfx1250, wave32, WMMA f32<=f16 16x16x32)
//
// Exact algebraic refactor: K/V per-edge projections split into node part
// (GEMM, i-independent) + 16-dim relative part folded into per-edge dots.
// All dense GEMMs run on v_wmma_f32_16x16x32_f16 with 32x32 C per wave (16x16 for attn @ V).
// ---------------------------------------------------------------------------

typedef __attribute__((ext_vector_type(16))) _Float16 v16h;
typedef __attribute__((ext_vector_type(8)))  _Float16 h8v;
typedef __attribute__((ext_vector_type(8)))  float    v8f;

#define BB   4
#define NN   256
#define FD   256
#define RELD 16
#define HH   8
#define DD   32         // head dim
#define LDS_STRIDE 40   // halves per LDS tile row (80B: 16B-aligned, bank-skewed)

// ---------------------------------------------------------------------------
// Templated WMMA GEMM: C[M,N] = A[M,K] (f16, row-major) x op(B) (+bias)(+res)
//   bIsKN == 0 : B stored [N,K] row-major (compute A @ B^T)
//   bIsKN == 1 : B stored [K,N] row-major (transposed while staging to LDS)
// Batched via blockIdx.z: off(z) = (z/cnt)*outer + (z%cnt)*inner
// 128 threads = 4 waves in a 2x2 grid; block tile BMxBN; wave tile WTMxWTN;
// each wave holds RM*RN f32 accumulators and issues RM*RN WMMAs per K-chunk.
// ---------------------------------------------------------------------------
template<int BM, int BN, int WTM, int WTN>
__global__ __launch_bounds__(128)
void gemm_wmma(const _Float16* __restrict__ A, long aOuter, long aInner, int aCnt, int lda,
               const _Float16* __restrict__ B, long bOuter, long bInner, int bCnt, int ldb, int bIsKN,
               float* __restrict__ Cf, _Float16* __restrict__ Ch,
               long cOuter, long cInner, int cCnt, int ldc,
               const float* __restrict__ bias, const float* __restrict__ residual, int ldres,
               int K)
{
    constexpr int RM = WTM / 16;
    constexpr int RN = WTN / 16;

    __shared__ _Float16 As[BM * LDS_STRIDE];
    __shared__ _Float16 Bs[BN * LDS_STRIDE];

    const int z = blockIdx.z;
    const _Float16* Ab = A + (long)(z / aCnt) * aOuter + (long)(z % aCnt) * aInner;
    const _Float16* Bb = B + (long)(z / bCnt) * bOuter + (long)(z % bCnt) * bInner;
    const long coff    =     (long)(z / cCnt) * cOuter + (long)(z % cCnt) * cInner;

    const int m0 = blockIdx.y * BM;
    const int n0 = blockIdx.x * BN;
    const int t  = threadIdx.x;
    const int lane = t & 31;
    const int w    = t >> 5;
    const int wm   = (w >> 1) * WTM;     // wave's M offset within block tile
    const int wn   = (w & 1)  * WTN;     // wave's N offset
    const int fr   = lane & 15;          // fragment row/col within 16
    const int kb   = (lane >> 4) * 8;    // K sub-block base per ISA A/B layout

    v8f acc[RM][RN] = {};

    for (int kc = 0; kc < K; kc += 32) {
        // ---- stage A tile [BM rows x 32 halves] (8-half = 16B chunks) ----
        for (int cid = t; cid < BM * 4; cid += 128) {
            const int r = cid >> 2, c = cid & 3;
            const _Float16* gp = &Ab[(long)(m0 + r) * lda + kc + c * 8];
            *(h8v*)&As[r * LDS_STRIDE + c * 8] = *(const h8v*)gp;
            if (kc + 32 < K) __builtin_prefetch(gp + 32, 0, 0);
        }
        // ---- stage B tile as [n][k] ----
        if (bIsKN) {
            for (int cid = t; cid < 32 * (BN / 8); cid += 128) {
                const int k = cid / (BN / 8), c = cid % (BN / 8);
                const _Float16* gp = &Bb[(long)(kc + k) * ldb + n0 + c * 8];
                h8v bv = *(const h8v*)gp;
                if (kc + 32 < K) __builtin_prefetch(gp + (long)32 * ldb, 0, 0);
                #pragma unroll
                for (int e = 0; e < 8; ++e)
                    Bs[(c * 8 + e) * LDS_STRIDE + k] = bv[e];
            }
        } else {
            for (int cid = t; cid < BN * 4; cid += 128) {
                const int r = cid >> 2, c = cid & 3;
                const _Float16* gp = &Bb[(long)(n0 + r) * ldb + kc + c * 8];
                *(h8v*)&Bs[r * LDS_STRIDE + c * 8] = *(const h8v*)gp;
                if (kc + 32 < K) __builtin_prefetch(gp + 32, 0, 0);
            }
        }
        __syncthreads();

        // ---- register fragments per ISA 16-bit A/B layout ----
        // lane L<16: row L, K = [kb..kb+7] U [16+kb..16+kb+7]; lane L>=16: kb=8.
        v16h af[RM], bf[RN];
        #pragma unroll
        for (int rm = 0; rm < RM; ++rm) {
            const _Float16* p = &As[(wm + rm * 16 + fr) * LDS_STRIDE + kb];
            #pragma unroll
            for (int e = 0; e < 8; ++e) af[rm][e]     = p[e];
            #pragma unroll
            for (int e = 0; e < 8; ++e) af[rm][8 + e] = p[16 + e];
        }
        #pragma unroll
        for (int rn = 0; rn < RN; ++rn) {
            const _Float16* q = &Bs[(wn + rn * 16 + fr) * LDS_STRIDE + kb];
            #pragma unroll
            for (int e = 0; e < 8; ++e) bf[rn][e]     = q[e];
            #pragma unroll
            for (int e = 0; e < 8; ++e) bf[rn][8 + e] = q[16 + e];
        }
        // ---- WMMA burst: RM*RN matrix ops per synchronization ----
        #pragma unroll
        for (int rm = 0; rm < RM; ++rm)
            #pragma unroll
            for (int rn = 0; rn < RN; ++rn)
                acc[rm][rn] = __builtin_amdgcn_wmma_f32_16x16x32_f16(
                    false, af[rm], false, bf[rn], (short)0, acc[rm][rn], false, false);
        __syncthreads();
    }

    // ---- epilogue: C layout = VGPR v, lanes0-15 -> M=v, lanes16-31 -> M=8+v ----
    #pragma unroll
    for (int rm = 0; rm < RM; ++rm) {
        #pragma unroll
        for (int rn = 0; rn < RN; ++rn) {
            const int nCol  = n0 + wn + rn * 16 + (lane & 15);
            const int mBase = m0 + wm + rm * 16 + ((lane >> 4) ? 8 : 0);
            #pragma unroll
            for (int v = 0; v < 8; ++v) {
                const int mRow = mBase + v;
                float val = acc[rm][rn][v];
                if (bias)     val += bias[nCol];
                if (residual) val += residual[(long)mRow * ldres + nCol];
                const long idx = coff + (long)mRow * ldc + nCol;
                if (Cf) Cf[idx] = val;
                if (Ch) Ch[idx] = (_Float16)val;
            }
        }
    }
}

// ---------------------------------------------------------------------------
// Pack: f32 -> f16 for node features; transpose W's to [out,in]; split rel rows.
// ---------------------------------------------------------------------------
__global__ __launch_bounds__(256)
void pack_all(const float* __restrict__ nf,
              const float* __restrict__ Wq, const float* __restrict__ Wk,
              const float* __restrict__ Wv, const float* __restrict__ Wo,
              _Float16* nfh, _Float16* WqT, _Float16* WkT, _Float16* WvT,
              _Float16* WoT, _Float16* WkR, _Float16* WvR)
{
    int idx = blockIdx.x * 256 + threadIdx.x;
    const int NFE = BB * NN * FD;   // 262144
    const int WE  = FD * FD;        // 65536
    const int RE  = RELD * FD;      // 4096
    if (idx < NFE) { nfh[idx] = (_Float16)nf[idx]; return; }
    idx -= NFE;
    if (idx < WE) { int o = idx >> 8, i = idx & 255; WqT[o*FD+i] = (_Float16)Wq[i*FD+o]; return; }
    idx -= WE;
    if (idx < WE) { int o = idx >> 8, i = idx & 255; WkT[o*FD+i] = (_Float16)Wk[i*FD+o]; return; }
    idx -= WE;
    if (idx < WE) { int o = idx >> 8, i = idx & 255; WvT[o*FD+i] = (_Float16)Wv[i*FD+o]; return; }
    idx -= WE;
    if (idx < WE) { int o = idx >> 8, i = idx & 255; WoT[o*FD+i] = (_Float16)Wo[i*FD+o]; return; }
    idx -= WE;
    if (idx < RE) { int r = idx >> 8, o = idx & 255; WkR[r*FD+o] = (_Float16)Wk[(FD+r)*FD+o]; return; }
    idx -= RE;
    if (idx < RE) { int r = idx >> 8, o = idx & 255; WvR[r*FD+o] = (_Float16)Wv[(FD+r)*FD+o]; return; }
}

// ---------------------------------------------------------------------------
// tK[b,i,h,r] = sum_d WkR[r, h*32+d] * Q[b,i,h*32+d]   (folds Q . K_rel)
// ---------------------------------------------------------------------------
__global__ __launch_bounds__(128)
void k_tk(const _Float16* __restrict__ Qh, const _Float16* __restrict__ WkR,
          float* __restrict__ tK)
{
    const int bi = blockIdx.x;          // b*256 + i
    const int t  = threadIdx.x;         // 0..127  -> (h, r)
    const int h  = t >> 4, rr = t & 15;
    const _Float16* q = Qh  + (long)bi * FD + h * DD;
    const _Float16* w = WkR + rr * FD + h * DD;
    float s = 0.f;
    #pragma unroll
    for (int d = 0; d < DD; ++d) s += (float)q[d] * (float)w[d];
    tK[((long)bi * HH + h) * RELD + rr] = s;
}

// ---------------------------------------------------------------------------
// scores += rel.tK, scale, softmax over j, emit attn (f16). One block per row.
// ---------------------------------------------------------------------------
__global__ __launch_bounds__(256)
void k_softmax(const float* __restrict__ scores, const float* __restrict__ rel,
               const float* __restrict__ tK, _Float16* __restrict__ attn)
{
    const int row = blockIdx.x;          // (b*8 + h)*256 + i
    const int i   = row & 255;
    const int bh  = row >> 8;
    const int h   = bh & 7, b = bh >> 3;
    const int j   = threadIdx.x;

    const long bi = (long)(b * NN + i);
    const float* relp = rel + (bi * NN + j) * RELD;
    const float* tkp  = tK  + (bi * HH + h) * RELD;
    float s2 = 0.f;
    #pragma unroll
    for (int r = 0; r < RELD; ++r) s2 += relp[r] * tkp[r];

    float s = (scores[(long)row * NN + j] + s2) * 0.17677669529663689f; // 1/sqrt(32)

    __shared__ float red[8];
    // max-reduce (8 wave32s)
    float m = s;
    for (int off = 16; off; off >>= 1) m = fmaxf(m, __shfl_xor(m, off));
    if ((j & 31) == 0) red[j >> 5] = m;
    __syncthreads();
    if (j < 8) {
        float x = red[j];
        for (int off = 4; off; off >>= 1) x = fmaxf(x, __shfl_xor(x, off));
        if (j == 0) red[0] = x;
    }
    __syncthreads();
    m = red[0];
    float e = __expf(s - m);
    __syncthreads();
    // sum-reduce
    float sum = e;
    for (int off = 16; off; off >>= 1) sum += __shfl_xor(sum, off);
    if ((j & 31) == 0) red[j >> 5] = sum;
    __syncthreads();
    if (j < 8) {
        float x = red[j];
        for (int off = 4; off; off >>= 1) x += __shfl_xor(x, off);
        if (j == 0) red[0] = x;
    }
    __syncthreads();
    attn[(long)row * NN + j] = (_Float16)(e / red[0]);
}

// ---------------------------------------------------------------------------
// u[b,h,i,r] = sum_j attn*rel ; wvh = f16( wv + u @ WvR )   (folds attn.V_rel)
// ---------------------------------------------------------------------------
__global__ __launch_bounds__(256)
void k_uwv(const _Float16* __restrict__ attn, const float* __restrict__ rel,
           const _Float16* __restrict__ WvR, const float* __restrict__ wv,
           _Float16* __restrict__ wvh)
{
    __shared__ float u[HH * RELD];       // 128 values
    const int bi = blockIdx.x;           // b*256 + i
    const int b  = bi >> 8, i = bi & 255;
    const int t  = threadIdx.x;

    if (t < HH * RELD) {
        const int h = t >> 4, r = t & 15;
        const _Float16* ap = attn + ((long)((b * HH + h) * NN + i)) * NN;
        const float*    rp = rel  + ((long)bi * NN) * RELD + r;
        float s = 0.f;
        for (int j = 0; j < NN; ++j) s += (float)ap[j] * rp[(long)j * RELD];
        u[t] = s;
    }
    __syncthreads();

    const int h = t >> 5, d = t & 31;
    const int col = h * DD + d;
    float val = wv[(long)bi * FD + col];
    #pragma unroll
    for (int r = 0; r < RELD; ++r) val += u[h * RELD + r] * (float)WvR[r * FD + col];
    wvh[(long)bi * FD + col] = (_Float16)val;
}

// ---------------------------------------------------------------------------
extern "C" void kernel_launch(void* const* d_in, const int* in_sizes, int n_in,
                              void* d_out, int out_size, void* d_ws, size_t ws_size,
                              hipStream_t stream)
{
    const float* nf  = (const float*)d_in[0];
    const float* rel = (const float*)d_in[1];
    const float* Wq  = (const float*)d_in[2];
    const float* bq  = (const float*)d_in[3];
    const float* Wk  = (const float*)d_in[4];
    const float* bk  = (const float*)d_in[5];
    const float* Wv  = (const float*)d_in[6];
    const float* bv  = (const float*)d_in[7];
    const float* Wo  = (const float*)d_in[8];
    const float* bo  = (const float*)d_in[9];
    float* out = (float*)d_out;

    // workspace carve-up (all buffers fully written before read)
    char* p = (char*)d_ws;
    auto alloc = [&](size_t bytes) -> char* {
        char* q = p; p += (bytes + 255) & ~(size_t)255; return q;
    };
    const size_t MN = (size_t)BB * NN;               // 1024 rows
    _Float16* nfh  = (_Float16*)alloc(MN * FD * 2);
    _Float16* WqT  = (_Float16*)alloc(FD * FD * 2);
    _Float16* WkT  = (_Float16*)alloc(FD * FD * 2);
    _Float16* WvT  = (_Float16*)alloc(FD * FD * 2);
    _Float16* WoT  = (_Float16*)alloc(FD * FD * 2);
    _Float16* WkR  = (_Float16*)alloc(RELD * FD * 2);
    _Float16* WvR  = (_Float16*)alloc(RELD * FD * 2);
    _Float16* Qh   = (_Float16*)alloc(MN * FD * 2);
    _Float16* Knh  = (_Float16*)alloc(MN * FD * 2);
    _Float16* Vnh  = (_Float16*)alloc(MN * FD * 2);
    float*    tK   = (float*)   alloc(MN * HH * RELD * 4);
    float*    sc   = (float*)   alloc((size_t)BB * HH * NN * NN * 4);   // 8 MB
    _Float16* attn = (_Float16*)alloc((size_t)BB * HH * NN * NN * 2);   // 4 MB
    float*    wvf  = (float*)   alloc(MN * FD * 4);
    _Float16* wvh  = (_Float16*)alloc(MN * FD * 2);

    // 1) pack / transpose / f32->f16
    {
        const int total = BB*NN*FD + 4*FD*FD + 2*RELD*FD;   // 532480
        pack_all<<<(total + 255) / 256, 256, 0, stream>>>(
            nf, Wq, Wk, Wv, Wo, nfh, WqT, WkT, WvT, WoT, WkR, WvR);
    }

    // 2-4) node projections: Qn/Kn/Vn = nf @ W + b  (WMMA 64x64 tiles, out f16)
    {
        dim3 g(FD / 64, (int)(MN / 64), 1);
        gemm_wmma<64,64,32,32><<<g, 128, 0, stream>>>(
            nfh, 0,0,1, FD,  WqT, 0,0,1, FD, 0,
            nullptr, Qh,  0,0,1, FD,  bq, nullptr, 0, FD);
        gemm_wmma<64,64,32,32><<<g, 128, 0, stream>>>(
            nfh, 0,0,1, FD,  WkT, 0,0,1, FD, 0,
            nullptr, Knh, 0,0,1, FD,  bk, nullptr, 0, FD);
        gemm_wmma<64,64,32,32><<<g, 128, 0, stream>>>(
            nfh, 0,0,1, FD,  WvT, 0,0,1, FD, 0,
            nullptr, Vnh, 0,0,1, FD,  bv, nullptr, 0, FD);
    }

    // 5) tK (rel-part of Q.K folded to 16 dims)
    k_tk<<<(int)MN, 128, 0, stream>>>(Qh, WkR, tK);

    // 6) S1[b,h] = Q_bh @ Kn_bh^T  (batched WMMA, z = b*8+h, K=32)
    {
        dim3 g(NN / 64, NN / 64, BB * HH);
        gemm_wmma<64,64,32,32><<<g, 128, 0, stream>>>(
            Qh,  (long)NN*FD, DD, HH, FD,
            Knh, (long)NN*FD, DD, HH, FD, 0,
            sc, nullptr, (long)NN*NN, 0, 1, NN,
            nullptr, nullptr, 0, DD);
    }

    // 7) add rel-term, scale, softmax -> attn(f16)
    k_softmax<<<BB * HH * NN, 256, 0, stream>>>(sc, rel, tK, attn);

    // 8) wv1[b,h] = attn_bh @ Vn_bh  (batched WMMA, B stored [K,N], K=256, N=32)
    {
        dim3 g(DD / 32, NN / 32, BB * HH);
        gemm_wmma<32,32,16,16><<<g, 128, 0, stream>>>(
            attn, (long)NN*NN, 0, 1, NN,
            Vnh,  (long)NN*FD, DD, HH, FD, 1,
            wvf, nullptr, (long)NN*FD, DD, HH, FD,
            nullptr, nullptr, 0, NN);
    }

    // 9) add rel-part of attn@V, emit wvh (f16)
    k_uwv<<<(int)MN, 256, 0, stream>>>(attn, rel, WvR, wvf, wvh);

    // 10) out = nf + wvh @ Wo + bo  (WMMA, f32 out, fused residual+bias)
    {
        dim3 g(FD / 64, (int)(MN / 64), 1);
        gemm_wmma<64,64,32,32><<<g, 128, 0, stream>>>(
            wvh, 0,0,1, FD,  WoT, 0,0,1, FD, 0,
            out, nullptr, 0,0,1, FD,  bo, nf, FD, FD);
    }
}
